// MDASSA_6502580486253
// MI455X (gfx1250) — matrix operand order repaired and checked by reading, not compiled, not run on hardware
//
#include <hip/hip_runtime.h>

#define NWIN  4096
#define NTOK  64
#define DIM   192
#define HEADS 6
#define HD    32
#define QKVN  576   // 3*DIM

typedef __attribute__((ext_vector_type(16))) _Float16 v16h;
typedef __attribute__((ext_vector_type(8)))  _Float16 v8h;
typedef __attribute__((ext_vector_type(8)))  float    v8f;
typedef __attribute__((ext_vector_type(4)))  unsigned int u32x4;
typedef __attribute__((ext_vector_type(8)))  unsigned int u32x8;

// ---- A-operand load from row-major f16 LDS tile (16 rows x 32 K) ----
// Layout: lane m=lane&15; lanes<16 -> K base 0, lanes>=16 -> K base 16;
// halves 0..15 = K kb+0..15 (contiguous, matches B operand).
__device__ __forceinline__ v16h load_a_tile(const _Float16* base, int ld, int lane) {
  const int m  = lane & 15;
  const int kb = (lane < 16) ? 0 : 16;
  const _Float16* p = base + m * ld + kb;
  v8h lo = *(const v8h*)(p);
  v8h hi = *(const v8h*)(p + 8);
  v16h a;
#pragma unroll
  for (int i = 0; i < 8; ++i) { a[i] = lo[i]; a[i + 8] = hi[i]; }
  return a;
}

__device__ __forceinline__ v8f wmma_f16(v16h a, v16h b, v8f c) {
  return __builtin_amdgcn_wmma_f32_16x16x32_f16(false, a, false, b, (short)0, c,
                                                false, false);
}

// ---- prep: transpose+convert weights to N-major f16, fold q-scale, gather bias ----
__global__ __launch_bounds__(256) void prep_kernel(
    const float* __restrict__ Wq,  const float* __restrict__ bq,
    const float* __restrict__ Wkv, const float* __restrict__ bkv,
    const float* __restrict__ bias_table, const float* __restrict__ Wp,
    const int* __restrict__ rel_index,
    _Float16* __restrict__ wqkvt, _Float16* __restrict__ wpt,
    float* __restrict__ bqkv, float* __restrict__ biasg) {
  const float scale = 0.17677669529663687f;  // 32^-0.5 folded into Wq/bq
  int i = blockIdx.x * 256 + threadIdx.x;
  if (i < QKVN * DIM) {
    int n = i / DIM, k = i % DIM;
    float v = (n < DIM) ? Wq[k * DIM + n] * scale : Wkv[k * (2 * DIM) + (n - DIM)];
    wqkvt[i] = (_Float16)v;
  }
  if (i < DIM * DIM) {
    int n = i / DIM, k = i % DIM;
    wpt[i] = (_Float16)Wp[k * DIM + n];
  }
  if (i < QKVN) bqkv[i] = (i < DIM) ? bq[i] * scale : bkv[i - DIM];
  if (i < HEADS * NTOK * NTOK) {
    int h = i / (NTOK * NTOK), q = (i / NTOK) % NTOK, kc = i % NTOK;
    biasg[i] = bias_table[rel_index[q * NTOK + kc] * HEADS + h];
  }
}

// ---- fused window attention: one workgroup (8 wave32) per window ----
__global__ __launch_bounds__(256, 1) void attn_kernel(
    const float* __restrict__ x, const float* __restrict__ bp,
    const float* __restrict__ w,
    const _Float16* __restrict__ wqkvt, const _Float16* __restrict__ wpt,
    const float* __restrict__ bqkv, const float* __restrict__ biasg,
    float* __restrict__ out) {
  __shared__ alignas(32) _Float16 smem[86016];   // 168 KB of the 320 KB WGP LDS
  _Float16* xs  = smem;            // [64][192] f16 x
  _Float16* qs  = smem + 12288;    // [64][192] q (scale folded)
  _Float16* ks  = smem + 24576;    // [64][192] k
  _Float16* vsT = smem + 36864;    // [192][64] v transposed (B-operand ready)
  _Float16* as_ = smem + 49152;    // [6][64][64] blended attention probs
  _Float16* hs  = smem + 73728;    // [64][192] concat head outputs
  float*    os  = (float*)as_;     // [64][192] f32 output staging  (reuses as_)
  float*    stg = (float*)as_;     // [64][192] f32 x staging (TDM)  (reuses as_)

  const int tid  = threadIdx.x;
  const int wave = tid >> 5;
  const int lane = tid & 31;
  const int lm   = lane & 15;
  const int mo   = (lane < 16) ? 0 : 8;   // C/D row-half offset
  const int ko   = (lane < 16) ? 0 : 16;  // B-operand K offset

  float e0 = __expf(w[0]), e1 = __expf(w[1]);
  const float ww0 = e0 / (e0 + e1);
  const float ww1 = e1 / (e0 + e1);

  // warm weight region in cache hierarchy (global_prefetch_b8)
  {
    const char* wb = (const char*)wqkvt;
    for (int off = tid * 64; off < (QKVN * DIM + DIM * DIM) * 2; off += 256 * 64)
      __builtin_prefetch(wb + off, 0, 0);
  }

  // ---- Stage A: Tensor Data Mover pulls the fp32 x tile (48 KB) into LDS ----
  const float* xg = x + (size_t)blockIdx.x * (NTOK * DIM);
  if (wave == 0) {
    const unsigned lds_off = (unsigned)(unsigned long long)(uintptr_t)stg;
    const unsigned long long ga = (unsigned long long)(uintptr_t)xg;
    u32x4 g0;
    g0[0] = 1u;                                          // count=1 valid descriptor
    g0[1] = lds_off;                                     // lds_addr
    g0[2] = (unsigned)ga;                                // global_addr[31:0]
    g0[3] = (unsigned)((ga >> 32) & 0x1FFFFFFu) | (2u << 30);  // addr[56:32], type=2
    u32x8 g1;
    g1[0] = (2u << 16);        // workgroup_mask=0, data_size=2 (4-byte elements)
    g1[1] = (12288u << 16);    // tensor_dim0[15:0] = 12288 (bits 63:48)
    g1[2] = (1u << 16);        // tensor_dim0 hi=0, tensor_dim1[15:0]=1
    g1[3] = (12288u << 16);    // tensor_dim1 hi=0, tile_dim0=12288
    g1[4] = 1u;                // tile_dim1=1, tile_dim2=0
    g1[5] = 12288u;            // tensor_dim0_stride[31:0]
    g1[6] = 0u;
    g1[7] = 0u;
    u32x4 gz = {0u, 0u, 0u, 0u};
    asm volatile("tensor_load_to_lds %0, %1, %2, %3"
                 :: "s"(g0), "s"(g1), "s"(gz), "s"(gz) : "memory");
    __builtin_amdgcn_s_wait_tensorcnt(0);
  }
  __syncthreads();

  // fp32 (LDS staging) -> fp16 operand buffer
  for (int i = tid * 4; i < NTOK * DIM; i += 256 * 4) {
    float4 f = *(const float4*)(stg + i);
    xs[i + 0] = (_Float16)f.x; xs[i + 1] = (_Float16)f.y;
    xs[i + 2] = (_Float16)f.z; xs[i + 3] = (_Float16)f.w;
  }
  __syncthreads();

  // ---- Stage B: QKV = x @ [Wq*scale | Wk | Wv] + b   (M=64, N=576, K=192) ----
  {
    const int m0 = (wave >> 1) * 16;
    v16h A[6];
#pragma unroll
    for (int kk = 0; kk < 6; ++kk) A[kk] = load_a_tile(xs + m0 * DIM + kk * 32, DIM, lane);
    const int nh = wave & 1;
    for (int j = 0; j < 18; ++j) {
      const int nt = nh * 18 + j;
      const int ncol = nt * 16 + lm;
      const float bb = bqkv[ncol];
      v8f c;
#pragma unroll
      for (int r = 0; r < 8; ++r) c[r] = bb;
#pragma unroll
      for (int kk = 0; kk < 6; ++kk) {
        v16h B = *(const v16h*)(wqkvt + ncol * DIM + kk * 32 + ko);
        c = wmma_f16(A[kk], B, c);
      }
      // region is tile-uniform (16-col tiles never straddle the 192 boundaries):
      // force a scalar branch instead of per-lane exec masking.
      const int reg = __builtin_amdgcn_readfirstlane(nt >= 24 ? 2 : (nt >= 12 ? 1 : 0));
      if (reg == 0) {
#pragma unroll
        for (int r = 0; r < 8; ++r) qs[(m0 + mo + r) * DIM + ncol] = (_Float16)c[r];
      } else if (reg == 1) {
        const int kc = ncol - DIM;
#pragma unroll
        for (int r = 0; r < 8; ++r) ks[(m0 + mo + r) * DIM + kc] = (_Float16)c[r];
      } else {
        const int d = ncol - 2 * DIM;
#pragma unroll
        for (int r = 0; r < 8; ++r) vsT[d * NTOK + (m0 + mo + r)] = (_Float16)c[r];
      }
    }
  }
  __syncthreads();

  // ---- Stage C: scores = q k^T + bias, then blend ww0*softmax + ww1*relu^2 ----
  // Each wave owns complete rows -> softmax is an in-wave 16-lane reduction.
#pragma unroll
  for (int p = 0; p < 3; ++p) {
    const int idx = wave * 3 + p;          // 24 (head, m-tile) pairs
    const int h  = idx >> 2;
    const int m0 = (idx & 3) * 16;
    v16h Aq = load_a_tile(qs + m0 * DIM + h * HD, DIM, lane);
    v8f acc[4];
#pragma unroll
    for (int nt = 0; nt < 4; ++nt) {
      const int ntok = nt * 16 + lm;
      v8f c;
#pragma unroll
      for (int r = 0; r < 8; ++r) c[r] = biasg[h * 4096 + (m0 + mo + r) * NTOK + ntok];
      v16h Bk = *(const v16h*)(ks + ntok * DIM + h * HD + ko);
      acc[nt] = wmma_f16(Aq, Bk, c);
    }
#pragma unroll
    for (int r = 0; r < 8; ++r) {
      float mx = acc[0][r];
#pragma unroll
      for (int nt = 1; nt < 4; ++nt) mx = fmaxf(mx, acc[nt][r]);
#pragma unroll
      for (int msk = 8; msk >= 1; msk >>= 1) mx = fmaxf(mx, __shfl_xor(mx, msk, 32));
      float e[4], s = 0.f;
#pragma unroll
      for (int nt = 0; nt < 4; ++nt) { e[nt] = __expf(acc[nt][r] - mx); s += e[nt]; }
#pragma unroll
      for (int msk = 8; msk >= 1; msk >>= 1) s += __shfl_xor(s, msk, 32);
      const float inv = ww0 / s;
      const int m = m0 + mo + r;
#pragma unroll
      for (int nt = 0; nt < 4; ++nt) {
        const float t  = acc[nt][r];
        const float rq = t > 0.f ? t * t : 0.f;
        as_[h * 4096 + m * NTOK + nt * 16 + lm] = (_Float16)(e[nt] * inv + ww1 * rq);
      }
    }
  }
  __syncthreads();

  // ---- Stage D: per-head  out_h = a_h @ v_h   (M=64, N=32, K=64) ----
#pragma unroll
  for (int i = 0; i < 6; ++i) {
    const int t = wave * 6 + i;            // 48 tiles
    const int h = t >> 3;
    const int m0 = ((t & 7) >> 1) * 16;
    const int dglob = h * HD + (t & 1) * 16 + lm;
    v8f c;
#pragma unroll
    for (int r = 0; r < 8; ++r) c[r] = 0.f;
#pragma unroll
    for (int kk = 0; kk < 2; ++kk) {
      v16h Aa = load_a_tile(as_ + h * 4096 + m0 * NTOK + kk * 32, NTOK, lane);
      v16h Bv = *(const v16h*)(vsT + dglob * NTOK + kk * 32 + ko);
      c = wmma_f16(Aa, Bv, c);
    }
#pragma unroll
    for (int r = 0; r < 8; ++r) hs[(m0 + mo + r) * DIM + dglob] = (_Float16)c[r];
  }
  __syncthreads();

  // ---- Stage E: out = hs @ Wp + bp  (M=64, N=192, K=192) ----
  {
    const int m0 = (wave >> 1) * 16;
    v16h A[6];
#pragma unroll
    for (int kk = 0; kk < 6; ++kk) A[kk] = load_a_tile(hs + m0 * DIM + kk * 32, DIM, lane);
#pragma unroll
    for (int i = 0; i < 6; ++i) {
      const int ncol = ((wave & 1) * 6 + i) * 16 + lm;
      const float bb = bp[ncol];
      v8f c;
#pragma unroll
      for (int r = 0; r < 8; ++r) c[r] = bb;
#pragma unroll
      for (int kk = 0; kk < 6; ++kk) {
        v16h B = *(const v16h*)(wpt + ncol * DIM + kk * 32 + ko);
        c = wmma_f16(A[kk], B, c);
      }
#pragma unroll
      for (int r = 0; r < 8; ++r) os[(m0 + mo + r) * DIM + ncol] = c[r];
    }
  }
  __syncthreads();

  // coalesced fp32 store
  float* og = out + (size_t)blockIdx.x * (NTOK * DIM);
  for (int i = tid * 4; i < NTOK * DIM; i += 256 * 4)
    *(float4*)(og + i) = *(const float4*)(os + i);
}

extern "C" void kernel_launch(void* const* d_in, const int* in_sizes, int n_in,
                              void* d_out, int out_size, void* d_ws, size_t ws_size,
                              hipStream_t stream) {
  const float* x          = (const float*)d_in[0];
  const float* Wq         = (const float*)d_in[1];
  const float* bq         = (const float*)d_in[2];
  const float* Wkv        = (const float*)d_in[3];
  const float* bkv        = (const float*)d_in[4];
  const float* bias_table = (const float*)d_in[5];
  const float* Wp         = (const float*)d_in[6];
  const float* bp         = (const float*)d_in[7];
  const float* w          = (const float*)d_in[8];
  const int*   rel_index  = (const int*)d_in[9];

  char* ws = (char*)d_ws;
  _Float16* wqkvt = (_Float16*)(ws);            // 576*192 f16 = 221184 B
  _Float16* wpt   = (_Float16*)(ws + 221184);   // 192*192 f16 =  73728 B
  float*    bqkv  = (float*)(ws + 294912);      // 576 f32
  float*    biasg = (float*)(ws + 297216);      // 6*64*64 f32

  prep_kernel<<<(QKVN * DIM + 255) / 256, 256, 0, stream>>>(
      Wq, bq, Wkv, bkv, bias_table, Wp, rel_index, wqkvt, wpt, bqkv, biasg);
  attn_kernel<<<NWIN, 256, 0, stream>>>(x, bp, w, wqkvt, wpt, bqkv, biasg,
                                        (float*)d_out);
}
